// Block_47811575939457
// MI455X (gfx1250) — matrix operand order repaired and checked by reading, not compiled, not run on hardware
//
#include <hip/hip_runtime.h>

// Problem constants (B=2, T=2048, E=768, H=12, HS=64)
#define BB   2
#define TT   2048
#define BT   4096      // B*T
#define EE   768
#define HH   12
#define HS   64
#define FF   3072      // 4*E
#define QKVN 2304      // 3*E

typedef __attribute__((ext_vector_type(16))) __bf16       v16bf;
typedef __attribute__((ext_vector_type(8)))  float        v8f;
typedef __attribute__((ext_vector_type(8)))  unsigned int v8u;
typedef __attribute__((ext_vector_type(4)))  unsigned int v4u;
typedef __attribute__((ext_vector_type(8)))  int          v8i;
typedef __attribute__((ext_vector_type(4)))  int          v4i;

#if __has_builtin(__builtin_amdgcn_tensor_load_to_lds)
#define HAVE_TDM 1
#else
#define HAVE_TDM 0
#endif

__device__ __forceinline__ unsigned short f2bf(float f) {
  unsigned int u = __builtin_bit_cast(unsigned int, f);
  unsigned int r = u + 0x7FFFu + ((u >> 16) & 1u);   // round-to-nearest-even
  return (unsigned short)(r >> 16);
}

__device__ __forceinline__ float shfl_xor_f(float v, int mask) {
  // wave32: ds_bpermute uses byte index, bits [6:2]
  int lane = (int)(threadIdx.x & 31);
  int src = ((lane ^ mask) << 2);
  return __builtin_bit_cast(float, __builtin_amdgcn_ds_bpermute(src, __builtin_bit_cast(int, v)));
}

// ---------------------------------------------------------------------------
// LayerNorm (f32 in) -> bf16 out.  One 256-thread block per row of 768.
// ---------------------------------------------------------------------------
__global__ __launch_bounds__(256)
void ln_bf16_kernel(const float* __restrict__ x, const float* __restrict__ g,
                    const float* __restrict__ b, unsigned short* __restrict__ out) {
  __shared__ float red[8];
  __shared__ float stat[2];
  int row  = blockIdx.x;
  int t    = threadIdx.x;
  int wid  = t >> 5, lane = t & 31;
  const float* xr = x + (size_t)row * EE;
  float v0 = xr[t], v1 = xr[t + 256], v2 = xr[t + 512];

  float s = v0 + v1 + v2;
  for (int m = 1; m < 32; m <<= 1) s += shfl_xor_f(s, m);
  if (lane == 0) red[wid] = s;
  __syncthreads();
  if (t == 0) { float tot = 0.f; for (int i = 0; i < 8; ++i) tot += red[i]; stat[0] = tot * (1.0f / EE); }
  __syncthreads();
  float mean = stat[0];
  float d0 = v0 - mean, d1 = v1 - mean, d2 = v2 - mean;
  float q = d0 * d0 + d1 * d1 + d2 * d2;
  for (int m = 1; m < 32; m <<= 1) q += shfl_xor_f(q, m);
  __syncthreads();
  if (lane == 0) red[wid] = q;
  __syncthreads();
  if (t == 0) { float tot = 0.f; for (int i = 0; i < 8; ++i) tot += red[i]; stat[1] = rsqrtf(tot * (1.0f / EE) + 1e-5f); }
  __syncthreads();
  float rs = stat[1];
  size_t o = (size_t)row * EE;
  out[o + t      ] = f2bf(d0 * rs * g[t      ] + b[t      ]);
  out[o + t + 256] = f2bf(d1 * rs * g[t + 256] + b[t + 256]);
  out[o + t + 512] = f2bf(d2 * rs * g[t + 512] + b[t + 512]);
}

// ---------------------------------------------------------------------------
// Pack per-head Wq/Wk/Wv [H,E,HS] f32 into one bf16 [E, 3E] row-major matrix.
// ---------------------------------------------------------------------------
__global__ __launch_bounds__(256)
void pack_qkv_kernel(const float* __restrict__ Wq, const float* __restrict__ Wk,
                     const float* __restrict__ Wv, unsigned short* __restrict__ out) {
  int idx = blockIdx.x * 256 + threadIdx.x;      // over EE*QKVN
  int j   = idx % QKVN;
  int c   = idx / QKVN;
  const float* src; int jj;
  if (j < EE)            { src = Wq; jj = j;            }
  else if (j < 2 * EE)   { src = Wk; jj = j - EE;       }
  else                   { src = Wv; jj = j - 2 * EE;   }
  int h = jj >> 6, d = jj & 63;
  out[idx] = f2bf(src[((size_t)h * EE + c) * HS + d]);
}

__global__ __launch_bounds__(256)
void f32_to_bf16_kernel(const float* __restrict__ in, unsigned short* __restrict__ out, int n) {
  int i = blockIdx.x * 256 + threadIdx.x;
  if (i < n) out[i] = f2bf(in[i]);
}

// V slice of QKV [BT,2304] -> Vt [B][768][T]  (so P@V B-fragments are contiguous)
__global__ __launch_bounds__(256)
void transpose_v_kernel(const unsigned short* __restrict__ QKV, unsigned short* __restrict__ Vt) {
  int idx = blockIdx.x * 256 + threadIdx.x;      // over B*EE*TT
  int t   = idx % TT;
  int rem = idx / TT;
  int c   = rem % EE;
  int b   = rem / EE;
  Vt[idx] = QKV[(size_t)(b * TT + t) * QKVN + 2 * EE + c];
}

// ---------------------------------------------------------------------------
// bf16 GEMM, f32 accumulate:  C[M,N] = A[M,K] * B[K,N]  (+bias)(+ReLU)(+resid)
// 256 threads = 8 waves; block tile 128x64; wave tile 32x32 (2x2 WMMA).
// A-tile (128x32) staged into LDS by the Tensor Data Mover (D# descriptor,
// padded rows: 16 dwords data + 1 dword pad = the 17-dword LDS stride),
// tracked by TENSORcnt.  B-tile staged transposed by the threads.
// ---------------------------------------------------------------------------
__global__ __launch_bounds__(256)
void gemm_bf16_wmma(const unsigned short* __restrict__ A, const unsigned short* __restrict__ B,
                    int M, int N, int K,
                    const float* __restrict__ bias, const float* __restrict__ resid, int relu,
                    float* __restrict__ outF, unsigned short* __restrict__ outBF) {
  __shared__ unsigned int As[128 * 17];   // 128 rows x (16 dwords + 1 pad)
  __shared__ unsigned int Bs[64 * 17];    // 64 cols  x 32 bf16 (transposed) + pad

  int t = threadIdx.x;
  int wid = t >> 5, lane = t & 31;
  int g = lane >> 4, n16 = lane & 15;
  int blockM = blockIdx.y * 128, blockN = blockIdx.x * 64;
  int waveM = (wid >> 1) * 32, waveN = (wid & 1) * 32;

  v8f vzero = {0.f, 0.f, 0.f, 0.f, 0.f, 0.f, 0.f, 0.f};
  v8f c[2][2];
  c[0][0] = vzero; c[0][1] = vzero; c[1][0] = vzero; c[1][1] = vzero;

#if !HAVE_TDM
  // Fallback A staging: thread -> (row, 16-col half)
  int arow = t >> 1;
  int acol = (t & 1) * 16;
  const unsigned short* Aptr = A + (size_t)(blockM + arow) * K + acol;
#else
  unsigned int ldsA = (unsigned int)(unsigned long long)(&As[0]);
#endif
  // B staging: thread -> (k-pair, 4-col group)
  int kp = t & 15;            // k = 2*kp
  int bc = (t >> 4) * 4;      // col base within 64
  const unsigned short* Bptr = B + (size_t)(2 * kp) * N + blockN + bc;

  int nk = K >> 5;
  for (int kt = 0; kt < nk; ++kt) {
    __syncthreads();
#if HAVE_TDM
    if (wid == 0) {
      // ---- Tensor DMA: A tile [tile_dim1=128 rows][tile_dim0=32 elems] ----
      unsigned long long ga =
          (unsigned long long)(A + (size_t)blockM * K + (size_t)kt * 32);
      v4u g0;
      g0[0] = 1u;                                        // count=1, user D#
      g0[1] = ldsA;                                      // lds_addr
      g0[2] = (unsigned int)ga;                          // global_addr[31:0]
      g0[3] = (unsigned int)((ga >> 32) & 0x1FFFFFFull)  // global_addr[56:32]
              | (2u << 30);                              // type=2 (image)
      unsigned int td0 = (unsigned int)K;                // tensor_dim0
      unsigned int td1 = (unsigned int)M;                // tensor_dim1
      unsigned long long s0 = (unsigned long long)K;     // tensor_dim0_stride
      v8u g1;
      g1[0] = (1u << 16)        // data_size = 2 bytes
            | (1u << 20)        // pad_enable
            | (3u << 22)        // pad_interval: 16 dwords
            | (0u << 25);       // pad_amount: 1 dword
      g1[1] = (td0 & 0xFFFFu) << 16;                     // tensor_dim0[15:0]
      g1[2] = (td0 >> 16) | ((td1 & 0xFFFFu) << 16);     // td0[31:16], td1[15:0]
      g1[3] = (td1 >> 16) | (32u << 16);                 // td1[31:16], tile_dim0=32
      g1[4] = 128u;                                      // tile_dim1=128, tile_dim2=0
      g1[5] = (unsigned int)s0;                          // dim0_stride[31:0]
      g1[6] = (unsigned int)(s0 >> 32) & 0xFFFFu;        // dim0_stride[47:32]
      g1[7] = 0u;                                        // dim1_stride (unused, 2D)
      v4u gz = {0u, 0u, 0u, 0u};
#if defined(__clang_major__) && (__clang_major__ >= 23)
      v8u gz8 = {0u, 0u, 0u, 0u, 0u, 0u, 0u, 0u};
      __builtin_amdgcn_tensor_load_to_lds(g0, __builtin_bit_cast(v8i, g1),
                                          __builtin_bit_cast(v4i, gz),
                                          __builtin_bit_cast(v4i, gz),
                                          __builtin_bit_cast(v8i, gz8), 0);
#else
      __builtin_amdgcn_tensor_load_to_lds(g0, __builtin_bit_cast(v8i, g1),
                                          __builtin_bit_cast(v4i, gz),
                                          __builtin_bit_cast(v4i, gz), 0);
#endif
    }
#else
    { // manual A staging
      const unsigned short* ap = Aptr + kt * 32;
      unsigned int* dst = &As[arow * 17 + (acol >> 1)];
#pragma unroll
      for (int i = 0; i < 8; ++i) {
        unsigned int lo = ap[2 * i], hi = ap[2 * i + 1];
        dst[i] = lo | (hi << 16);
      }
    }
#endif
    { // stage B (transposed: Bs[col][k])
      const unsigned short* bp = Bptr + (size_t)kt * 32 * N;
      if (kt + 1 < nk) __builtin_prefetch(bp + (size_t)32 * N, 0, 1);
#pragma unroll
      for (int i = 0; i < 4; ++i) {
        unsigned int lo = bp[i];
        unsigned int hi = bp[N + i];
        Bs[(bc + i) * 17 + kp] = lo | (hi << 16);
      }
    }
#if HAVE_TDM
    if (wid == 0) __builtin_amdgcn_s_wait_tensorcnt(0);
#endif
    __syncthreads();

    v16bf a[2], bf[2];
#pragma unroll
    for (int mi = 0; mi < 2; ++mi) {
      int row = waveM + mi * 16 + n16;
      const unsigned int* src = &As[row * 17];
      v8u tmp;
#pragma unroll
      for (int v = 0; v < 8; ++v) {
        int kk = (v < 4 ? v : v + 4) + 4 * g;   // = K0/2 per 16-bit A layout
        tmp[v] = src[kk];
      }
      a[mi] = __builtin_bit_cast(v16bf, tmp);
    }
#pragma unroll
    for (int ni = 0; ni < 2; ++ni) {
      int col = waveN + ni * 16 + n16;
      const unsigned int* src = &Bs[col * 17];
      v8u tmp;
#pragma unroll
      for (int v = 0; v < 8; ++v) tmp[v] = src[(v < 4 ? v : v + 4) + 4 * g];   // K0/2 per 16-bit A layout
      bf[ni] = __builtin_bit_cast(v16bf, tmp);
    }
#pragma unroll
    for (int mi = 0; mi < 2; ++mi)
#pragma unroll
      for (int ni = 0; ni < 2; ++ni)
        c[mi][ni] = __builtin_amdgcn_wmma_f32_16x16x32_bf16(
            false, a[mi], false, bf[ni], (short)0, c[mi][ni], false, false);
  }

  // epilogue
#pragma unroll
  for (int mi = 0; mi < 2; ++mi)
#pragma unroll
    for (int ni = 0; ni < 2; ++ni) {
      int col = blockN + waveN + ni * 16 + n16;
      float bv = bias ? bias[col] : 0.f;
#pragma unroll
      for (int r = 0; r < 8; ++r) {
        int row = blockM + waveM + mi * 16 + r + 8 * g;
        float val = c[mi][ni][r] + bv;
        if (relu) val = fmaxf(val, 0.f);
        size_t idx = (size_t)row * N + col;
        if (resid) val += resid[idx];
        if (outF)  outF[idx]  = val;
        if (outBF) outBF[idx] = f2bf(val);
      }
    }
}

// ---------------------------------------------------------------------------
// Flash-style causal attention. 128 threads = 4 waves; each wave owns a
// 16-query tile of one (b,h). Steps over 32 keys: 4 WMMAs for S, online
// softmax (16-lane reductions), P restaged via LDS, 4 WMMAs for P@V.
// ---------------------------------------------------------------------------
__global__ __launch_bounds__(128)
void attn_kernel(const unsigned short* __restrict__ QKV,   // [BT,2304]: Q@0, K@768
                 const unsigned short* __restrict__ Vt,    // [B][768][T]
                 unsigned short* __restrict__ O) {         // [BT,768]
  __shared__ unsigned short Pl[4][16 * 34];

  int t = threadIdx.x;
  int wid = t >> 5, lane = t & 31, g = lane >> 4, n16 = lane & 15;
  int gw = blockIdx.x * 4 + wid;
  int qt = gw & 127;          // 128 query tiles of 16 per (b,h)
  int bh = gw >> 7;
  int h = bh % HH, b = bh / HH;
  unsigned short* Pw = Pl[wid];

  const int LD = QKVN;
  v8f vzero = {0.f, 0.f, 0.f, 0.f, 0.f, 0.f, 0.f, 0.f};

  // Q fragments (16x64 -> two 16x32 A-frags)
  v16bf aq[2];
  {
    int qrow = qt * 16 + n16;
    const unsigned short* qp = QKV + (size_t)(b * TT + qrow) * LD + h * HS;
#pragma unroll
    for (int half = 0; half < 2; ++half) {
      v8u tmp;
#pragma unroll
      for (int v = 0; v < 8; ++v) {
        int k0 = (v < 4 ? 2 * v : 16 + 2 * (v - 4)) + 8 * g + half * 32;
        tmp[v] = (unsigned)qp[k0] | ((unsigned)qp[k0 + 1] << 16);
      }
      aq[half] = __builtin_bit_cast(v16bf, tmp);
    }
  }

  v8f o[4]; o[0] = vzero; o[1] = vzero; o[2] = vzero; o[3] = vzero;
  float m[8], l[8];
#pragma unroll
  for (int r = 0; r < 8; ++r) { m[r] = -1e30f; l[r] = 0.f; }

  const float scale = 0.03608439182435161f;   // 768^-0.5 (ref scales by E, not HS)
  int jEnd = (qt * 16 + 15) >> 5;

  for (int j = 0; j <= jEnd; ++j) {
    int kbase = j * 32;
    v8f s[2];
#pragma unroll
    for (int sj = 0; sj < 2; ++sj) {
      int key = kbase + sj * 16 + n16;
      const unsigned short* kp = QKV + (size_t)(b * TT + key) * LD + EE + h * HS;
      v8u t0, t1;
#pragma unroll
      for (int v = 0; v < 8; ++v) {
        int k0 = (v < 4 ? 2 * v : 16 + 2 * (v - 4)) + 8 * g;
        t0[v] = (unsigned)kp[k0]      | ((unsigned)kp[k0 + 1]      << 16);
        t1[v] = (unsigned)kp[32 + k0] | ((unsigned)kp[32 + k0 + 1] << 16);
      }
      v8f acc = __builtin_amdgcn_wmma_f32_16x16x32_bf16(
          false, aq[0], false, __builtin_bit_cast(v16bf, t0), (short)0, vzero, false, false);
      s[sj] = __builtin_amdgcn_wmma_f32_16x16x32_bf16(
          false, aq[1], false, __builtin_bit_cast(v16bf, t1), (short)0, acc, false, false);
    }

    bool needMask = (kbase + 31) > (qt * 16);
#pragma unroll
    for (int sj = 0; sj < 2; ++sj)
#pragma unroll
      for (int r = 0; r < 8; ++r) {
        float v = s[sj][r] * scale;
        if (needMask) {
          int row = qt * 16 + r + 8 * g;
          int col = kbase + sj * 16 + n16;
          if (col > row) v = -1e30f;
        }
        s[sj][r] = v;
      }

    // online softmax per row (rows live across 16 lanes of each half-wave)
    float corr[8];
#pragma unroll
    for (int r = 0; r < 8; ++r) {
      float mx = fmaxf(s[0][r], s[1][r]);
      for (int msk = 1; msk < 16; msk <<= 1) mx = fmaxf(mx, shfl_xor_f(mx, msk));
      float mnew = fmaxf(m[r], mx);
      float cr = __expf(m[r] - mnew);
      float p0 = __expf(s[0][r] - mnew);
      float p1 = __expf(s[1][r] - mnew);
      float sum = p0 + p1;
      for (int msk = 1; msk < 16; msk <<= 1) sum += shfl_xor_f(sum, msk);
      l[r] = l[r] * cr + sum;
      m[r] = mnew;
      corr[r] = cr;
      int prow = r + 8 * g;
      Pw[prow * 34 + n16]      = f2bf(p0);
      Pw[prow * 34 + 16 + n16] = f2bf(p1);
    }
    asm volatile("" ::: "memory");

    // reshape P (16x32) into an A-fragment
    v16bf pa;
    {
      v8u tmp;
#pragma unroll
      for (int v = 0; v < 8; ++v) {
        int k0 = (v < 4 ? 2 * v : 16 + 2 * (v - 4)) + 8 * g;
        tmp[v] = (unsigned)Pw[n16 * 34 + k0] | ((unsigned)Pw[n16 * 34 + k0 + 1] << 16);
      }
      pa = __builtin_bit_cast(v16bf, tmp);
    }
    asm volatile("" ::: "memory");

    // O = O*corr + P @ V   (V^T buffer makes K-pairs contiguous)
#pragma unroll
    for (int nt = 0; nt < 4; ++nt) {
#pragma unroll
      for (int r = 0; r < 8; ++r) o[nt][r] *= corr[r];
      const unsigned short* vp =
          Vt + ((size_t)b * EE + h * HS + nt * 16 + n16) * TT + kbase;
      v8u tv;
#pragma unroll
      for (int v = 0; v < 8; ++v) {
        int k0 = (v < 4 ? 2 * v : 16 + 2 * (v - 4)) + 8 * g;
        tv[v] = (unsigned)vp[k0] | ((unsigned)vp[k0 + 1] << 16);
      }
      o[nt] = __builtin_amdgcn_wmma_f32_16x16x32_bf16(
          false, pa, false, __builtin_bit_cast(v16bf, tv), (short)0, o[nt], false, false);
    }
  }

  // finalize: divide by row sums, store bf16 at [b,t, h*64+d]
#pragma unroll
  for (int r = 0; r < 8; ++r) l[r] = 1.0f / l[r];
#pragma unroll
  for (int nt = 0; nt < 4; ++nt)
#pragma unroll
    for (int r = 0; r < 8; ++r) {
      int row = qt * 16 + r + 8 * g;
      O[(size_t)(b * TT + row) * EE + h * HS + nt * 16 + n16] = f2bf(o[nt][r] * l[r]);
    }
}

// ---------------------------------------------------------------------------
extern "C" void kernel_launch(void* const* d_in, const int* in_sizes, int n_in,
                              void* d_out, int out_size, void* d_ws, size_t ws_size,
                              hipStream_t stream) {
  const float* x     = (const float*)d_in[0];
  const float* Wq    = (const float*)d_in[1];
  const float* Wk    = (const float*)d_in[2];
  const float* Wv    = (const float*)d_in[3];
  const float* Wproj = (const float*)d_in[4];
  const float* bproj = (const float*)d_in[5];
  const float* W1    = (const float*)d_in[6];
  const float* b1    = (const float*)d_in[7];
  const float* W2    = (const float*)d_in[8];
  const float* b2    = (const float*)d_in[9];
  const float* g1    = (const float*)d_in[10];
  const float* be1   = (const float*)d_in[11];
  const float* g2    = (const float*)d_in[12];
  const float* be2   = (const float*)d_in[13];

  char* w = (char*)d_ws;
  auto alloc = [&](size_t bytes) -> char* {
    char* p = w;
    w += (bytes + 255) & ~(size_t)255;
    return p;
  };
  unsigned short* hln1   = (unsigned short*)alloc((size_t)BT * EE * 2);
  unsigned short* Wqkv   = (unsigned short*)alloc((size_t)EE * QKVN * 2);
  unsigned short* QKVb   = (unsigned short*)alloc((size_t)BT * QKVN * 2);
  unsigned short* Vtb    = (unsigned short*)alloc((size_t)BB * EE * TT * 2);
  unsigned short* Obuf   = (unsigned short*)alloc((size_t)BT * EE * 2);
  unsigned short* Wprojb = (unsigned short*)alloc((size_t)EE * EE * 2);
  float*          x1     = (float*)         alloc((size_t)BT * EE * 4);
  unsigned short* hln2   = (unsigned short*)alloc((size_t)BT * EE * 2);
  unsigned short* W1b    = (unsigned short*)alloc((size_t)EE * FF * 2);
  unsigned short* ff1    = (unsigned short*)alloc((size_t)BT * FF * 2);
  unsigned short* W2b    = (unsigned short*)alloc((size_t)FF * EE * 2);

  // 1) h = ln1(x) in bf16
  ln_bf16_kernel<<<BT, 256, 0, stream>>>(x, g1, be1, hln1);
  // 2) pack Q/K/V weights into [E, 3E] bf16
  pack_qkv_kernel<<<(EE * QKVN) / 256, 256, 0, stream>>>(Wq, Wk, Wv, Wqkv);
  // 3) QKV = h @ Wqkv  -> bf16 [BT, 2304]
  gemm_bf16_wmma<<<dim3(QKVN / 64, BT / 128), 256, 0, stream>>>(
      hln1, Wqkv, BT, QKVN, EE, nullptr, nullptr, 0, nullptr, QKVb);
  // 4) V^T for attention B-fragments
  transpose_v_kernel<<<(BB * EE * TT) / 256, 256, 0, stream>>>(QKVb, Vtb);
  // 5) flash attention -> O bf16 [BT, 768]
  attn_kernel<<<(BB * HH * (TT / 16)) / 4, 128, 0, stream>>>(QKVb, Vtb, Obuf);
  // 6) x1 = x + O @ Wproj + bproj   (f32)
  f32_to_bf16_kernel<<<(EE * EE + 255) / 256, 256, 0, stream>>>(Wproj, Wprojb, EE * EE);
  gemm_bf16_wmma<<<dim3(EE / 64, BT / 128), 256, 0, stream>>>(
      Obuf, Wprojb, BT, EE, EE, bproj, x, 0, x1, nullptr);
  // 7) h2 = ln2(x1) bf16
  ln_bf16_kernel<<<BT, 256, 0, stream>>>(x1, g2, be2, hln2);
  // 8) MLP
  f32_to_bf16_kernel<<<(EE * FF + 255) / 256, 256, 0, stream>>>(W1, W1b, EE * FF);
  f32_to_bf16_kernel<<<(FF * EE + 255) / 256, 256, 0, stream>>>(W2, W2b, FF * EE);
  gemm_bf16_wmma<<<dim3(FF / 64, BT / 128), 256, 0, stream>>>(
      hln2, W1b, BT, FF, EE, b1, nullptr, 1, nullptr, ff1);
  gemm_bf16_wmma<<<dim3(EE / 64, BT / 128), 256, 0, stream>>>(
      ff1, W2b, BT, EE, FF, b2, x1, 0, (float*)d_out, nullptr);

  (void)in_sizes; (void)n_in; (void)out_size; (void)ws_size;
}
